// EfficientAttention_16655883174652
// MI455X (gfx1250) — compile-verified, matrix operand order checked by reading, not run on hardware
//
#include <hip/hip_runtime.h>
#include <hip/hip_bf16.h>

typedef __attribute__((ext_vector_type(16))) _Float16 v16h;
typedef __attribute__((ext_vector_type(8)))  float    v8f;
typedef __attribute__((ext_vector_type(4)))  _Float16 h4;
typedef __attribute__((ext_vector_type(4)))  unsigned int u32x4;
typedef __attribute__((ext_vector_type(8)))  unsigned int u32x8;

// ---------------------------------------------------------------------------
// 16-element f16 A/B fragment (16x16x32 WMMA) from one LDS row.
// gfx1250 16-bit A layout: lane = row%16, half = lane/16,
//   VGPR v in 0..3 holds K = half*8 + 2v,2v+1 ; v in 4..7 holds K = 16 + ...
// (compiler merges these into two ds_load_b128 — verified in round 1 asm)
// ---------------------------------------------------------------------------
__device__ __forceinline__ v16h frag_row(const _Float16* rowp, int half) {
    v16h f;
#pragma unroll
    for (int v = 0; v < 8; ++v) {
        int k = (v < 4) ? (half * 8 + v * 2) : (16 + half * 8 + (v - 4) * 2);
        f[2 * v]     = rowp[k];
        f[2 * v + 1] = rowp[k + 1];
    }
    return f;
}

// ---------------------------------------------------------------------------
// Tensor Data Mover: async 2D f16 tile load global -> LDS.
// Tile: tile_w contiguous elems x tile_h rows; row pitch in LDS is
// tile_w*2 + 16 bytes via TDM padding (pad_interval code 4 = 128B,
// pad_amount code 3 = 4 dwords) => matches the 72-half LDS pitch below.
// D# per CDNA5 ISA. Issued once per wave; TENSORcnt.
// ---------------------------------------------------------------------------
__device__ __forceinline__ void tdm_load_2d_f16(
    unsigned lds_off, const _Float16* gptr,
    unsigned tile_w, unsigned tile_h,
    unsigned tensor_w, unsigned tensor_h, unsigned row_stride)
{
    unsigned long long ga = (unsigned long long)(uintptr_t)gptr;
    u32x4 g0;
    g0[0] = 1u;                                           // count=1 valid user D#
    g0[1] = lds_off;                                      // lds_addr (bytes)
    g0[2] = (unsigned)(ga & 0xFFFFFFFFu);                 // global_addr[31:0]
    g0[3] = (unsigned)((ga >> 32) & 0x1FFFFFFu)           // global_addr[56:32]
          | (2u << 30);                                   // type = 2 (image)
    u32x8 g1;
    g1[0] = (1u << 16)                                    // data_size = 2 bytes
          | (1u << 20)                                    // pad_enable
          | (4u << 22)                                    // pad_interval: 128B
          | (3u << 25);                                   // pad_amount: 4 dwords
    g1[1] = (tensor_w & 0xFFFFu) << 16;                   // tensor_dim0[15:0]
    g1[2] = ((tensor_w >> 16) & 0xFFFFu)                  // tensor_dim0[31:16]
          | ((tensor_h & 0xFFFFu) << 16);                 // tensor_dim1[15:0]
    g1[3] = ((tensor_h >> 16) & 0xFFFFu)                  // tensor_dim1[31:16]
          | (tile_w << 16);                               // tile_dim0
    g1[4] = tile_h;                                       // tile_dim1 (tile_dim2=0)
    g1[5] = row_stride;                                   // tensor_dim0_stride[31:0]
    g1[6] = 0;                                            // stride hi / dim1_stride
    g1[7] = 0;
    asm volatile("tensor_load_to_lds %0, %1" :: "s"(g0), "s"(g1) : "memory");
}

// ---------------------------------------------------------------------------
// Elementwise f32 -> f16 conversion (x and weight matrices), float4 vectorized.
// ---------------------------------------------------------------------------
__global__ __launch_bounds__(256) void cvt_f32_f16(
    const float* __restrict__ in, _Float16* __restrict__ out, int n4)
{
    int i = blockIdx.x * 256 + threadIdx.x;
    if (i < n4) {
        float4 v = reinterpret_cast<const float4*>(in)[i];
        h4 hv = { (_Float16)v.x, (_Float16)v.y, (_Float16)v.z, (_Float16)v.w };
        reinterpret_cast<h4*>(out)[i] = hv;
    }
}

// ---------------------------------------------------------------------------
// C[M,N] = A[M,K] @ W[K,N] + bias[N], A/W in f16, out f32 or f16.
// Block: 256 threads = 8 waves. Tile 128x64, K-step 32.
// Wave (wr,wc) owns a 32x32 patch = 2x2 WMMA accumulators.
// ---------------------------------------------------------------------------
template <typename TO>
__global__ __launch_bounds__(256) void gemm_bias_f16wmma(
    const _Float16* __restrict__ A, const _Float16* __restrict__ W,
    const float* __restrict__ bias, TO* __restrict__ C,
    int M, int K, int N)
{
    constexpr int BM = 128, BN = 64, BK = 32, LDT = BK + 8; // 40 halves / row
    __shared__ _Float16 sA[BM * LDT];
    __shared__ _Float16 sB[BN * LDT];   // stored [n][k] = W transposed

    const int tid  = threadIdx.x;
    const int lane = tid & 31, w = tid >> 5;
    const int wr = w & 3, wc = w >> 2;
    const int m0 = blockIdx.x * BM, n0 = blockIdx.y * BN;
    const int half = lane >> 4, rl = lane & 15;

    v8f acc[2][2] = {};

    for (int k0 = 0; k0 < K; k0 += BK) {
        __syncthreads();
        // Stage A tile 128x32 halves, 8-byte copies.
#pragma unroll
        for (int i = 0; i < 4; ++i) {
            int c = tid + i * 256;
            int r = c >> 3, c4 = (c & 7) << 2;
            h4 hv = *reinterpret_cast<const h4*>(&A[(size_t)(m0 + r) * K + k0 + c4]);
            *reinterpret_cast<h4*>(&sA[r * LDT + c4]) = hv;
        }
        // Stage W tile 32x64 transposed -> sB[n][k].
#pragma unroll
        for (int i = 0; i < 2; ++i) {
            int c = tid + i * 256;
            int r = c >> 4, c4 = (c & 15) << 2;
            h4 hv = *reinterpret_cast<const h4*>(&W[(size_t)(k0 + r) * N + n0 + c4]);
            sB[(c4 + 0) * LDT + r] = hv[0];
            sB[(c4 + 1) * LDT + r] = hv[1];
            sB[(c4 + 2) * LDT + r] = hv[2];
            sB[(c4 + 3) * LDT + r] = hv[3];
        }
        __syncthreads();

        v16h a0 = frag_row(&sA[(wr * 32 + 0  + rl) * LDT], half);
        v16h a1 = frag_row(&sA[(wr * 32 + 16 + rl) * LDT], half);
        v16h b0 = frag_row(&sB[(wc * 32 + 0  + rl) * LDT], half);
        v16h b1 = frag_row(&sB[(wc * 32 + 16 + rl) * LDT], half);

        acc[0][0] = __builtin_amdgcn_wmma_f32_16x16x32_f16(false, a0, false, b0, (short)0, acc[0][0], false, false);
        acc[0][1] = __builtin_amdgcn_wmma_f32_16x16x32_f16(false, a0, false, b1, (short)0, acc[0][1], false, false);
        acc[1][0] = __builtin_amdgcn_wmma_f32_16x16x32_f16(false, a1, false, b0, (short)0, acc[1][0], false, false);
        acc[1][1] = __builtin_amdgcn_wmma_f32_16x16x32_f16(false, a1, false, b1, (short)0, acc[1][1], false, false);
    }

    // Epilogue: one base pointer per tile, constant r*N offsets fold into
    // the 24-bit instruction offset.
#pragma unroll
    for (int i = 0; i < 2; ++i) {
#pragma unroll
        for (int j = 0; j < 2; ++j) {
            int col  = n0 + wc * 32 + j * 16 + rl;
            int row0 = m0 + wr * 32 + i * 16 + half * 8;
            float bv = bias[col];
            TO* cp = &C[(size_t)row0 * N + col];
#pragma unroll
            for (int r = 0; r < 8; ++r)
                cp[(size_t)r * N] = (TO)(acc[i][j][r] + bv);
        }
    }
}

// ---------------------------------------------------------------------------
// LayerNorm over D=1024 (in place, f32) + emit pre-scaled f16 copy for
// attention (qh = 0.125 * normed q).
// ---------------------------------------------------------------------------
__global__ __launch_bounds__(256) void layernorm_emit(
    float* __restrict__ q, _Float16* __restrict__ qh,
    const float* __restrict__ g, const float* __restrict__ bta)
{
    constexpr int D = 1024;
    float* p = q + (size_t)blockIdx.x * D;
    _Float16* ph = qh + (size_t)blockIdx.x * D;
    float xv[4], s = 0.f, s2 = 0.f;
#pragma unroll
    for (int i = 0; i < 4; ++i) {
        xv[i] = p[threadIdx.x + i * 256];
        s += xv[i]; s2 += xv[i] * xv[i];
    }
#pragma unroll
    for (int off = 16; off; off >>= 1) {
        s  += __shfl_xor(s,  off, 32);
        s2 += __shfl_xor(s2, off, 32);
    }
    __shared__ float sh[2][8];
    __shared__ float mu_s, rs_s;
    int lane = threadIdx.x & 31, w = threadIdx.x >> 5;
    if (lane == 0) { sh[0][w] = s; sh[1][w] = s2; }
    __syncthreads();
    if (threadIdx.x == 0) {
        float t = 0.f, t2 = 0.f;
#pragma unroll
        for (int i = 0; i < 8; ++i) { t += sh[0][i]; t2 += sh[1][i]; }
        float mu  = t / D;
        float var = t2 / D - mu * mu;
        mu_s = mu; rs_s = rsqrtf(var + 1e-5f);
    }
    __syncthreads();
    float mu = mu_s, rs = rs_s;
#pragma unroll
    for (int i = 0; i < 4; ++i) {
        int c = threadIdx.x + i * 256;
        float y = (xv[i] - mu) * rs * g[c] + bta[c];
        p[c]  = y;
        ph[c] = (_Float16)(y * 0.125f);   // fold d^-0.5 for attention
    }
}

// ---------------------------------------------------------------------------
// Flash attention. Block = (64 query rows, one head); 128 threads = 4 waves,
// each wave owns 16 query rows. d=64, keys streamed in tiles of 64.
// Q/K tiles arrive via the Tensor Data Mover (async, TENSORcnt); V is staged
// transposed through VGPRs. Epilogue fuses the "+ qn" residual, emits f16.
// ---------------------------------------------------------------------------
__global__ __launch_bounds__(128) void flash_attn(
    const _Float16* __restrict__ Qh, const _Float16* __restrict__ Kh,
    const _Float16* __restrict__ Vh, const float* __restrict__ Qn,
    _Float16* __restrict__ Outh)
{
    constexpr int LDT = 72;                 // 144B pitch = TDM 128B row + 16B pad
    __shared__ _Float16 sQ [64 * LDT];
    __shared__ _Float16 sK [64 * LDT];
    __shared__ _Float16 sVt[64 * LDT];      // [d-col][key]
    __shared__ _Float16 sP [4][16 * LDT];   // per-wave P relayout scratch

    const int tid = threadIdx.x, lane = tid & 31, w = tid >> 5;
    const int half = lane >> 4, rl = lane & 15;
    const int b = blockIdx.y >> 4, h = blockIdx.y & 15;   // H = 16
    const size_t rowbase = (size_t)b * 2048 + (size_t)blockIdx.x * 64;
    const int colbase = h * 64;

    // Q tile via TDM (pre-scaled f16 source).
    if (w == 0) {
        tdm_load_2d_f16((unsigned)(uintptr_t)sQ,
                        Qh + rowbase * 1024 + colbase,
                        64, 64, 1024, 4096, 1024);
        __builtin_amdgcn_s_wait_tensorcnt(0);
    }
    __syncthreads();
    v16h aq0 = frag_row(&sQ[(w * 16 + rl) * LDT + 0],  half);
    v16h aq1 = frag_row(&sQ[(w * 16 + rl) * LDT + 32], half);

    v8f O[4] = {};
    float mrow[8], lrow[8];
#pragma unroll
    for (int r = 0; r < 8; ++r) { mrow[r] = -1e30f; lrow[r] = 0.f; }

    for (int kt = 0; kt < 2048; kt += 64) {
        __syncthreads();                    // previous tile fully consumed
        const size_t kbase = (size_t)b * 2048 + kt;
        // K tile via TDM (wave 0), V tile transposed via VGPRs (all waves).
        if (w == 0)
            tdm_load_2d_f16((unsigned)(uintptr_t)sK,
                            Kh + kbase * 1024 + colbase,
                            64, 64, 1024, 4096, 1024);
#pragma unroll
        for (int i = 0; i < 8; ++i) {
            int c = tid + i * 128;
            int r = c >> 4, c4 = (c & 15) << 2;
            h4 vv = *reinterpret_cast<const h4*>(
                &Vh[(kbase + r) * 1024 + colbase + c4]);
            sVt[(c4 + 0) * LDT + r] = vv[0];
            sVt[(c4 + 1) * LDT + r] = vv[1];
            sVt[(c4 + 2) * LDT + r] = vv[2];
            sVt[(c4 + 3) * LDT + r] = vv[3];
        }
        if (w == 0) __builtin_amdgcn_s_wait_tensorcnt(0);
        __syncthreads();

        // S = Q K^T for this wave's 16 rows x 64 keys (4 tiles of 16 keys).
        v8f S[4];
#pragma unroll
        for (int jj = 0; jj < 4; ++jj) {
            v8f s0 = {};
            v16h bk0 = frag_row(&sK[(jj * 16 + rl) * LDT + 0],  half);
            v16h bk1 = frag_row(&sK[(jj * 16 + rl) * LDT + 32], half);
            s0 = __builtin_amdgcn_wmma_f32_16x16x32_f16(false, aq0, false, bk0, (short)0, s0, false, false);
            s0 = __builtin_amdgcn_wmma_f32_16x16x32_f16(false, aq1, false, bk1, (short)0, s0, false, false);
            S[jj] = s0;
        }

        // Online softmax (row stats across 16-lane half-groups, wave32 shfl).
        float alpha[8], psum[8];
#pragma unroll
        for (int r = 0; r < 8; ++r) {
            float m = fmaxf(fmaxf(S[0][r], S[1][r]), fmaxf(S[2][r], S[3][r]));
#pragma unroll
            for (int off = 8; off; off >>= 1) m = fmaxf(m, __shfl_xor(m, off, 32));
            float mn = fmaxf(mrow[r], m);
            alpha[r] = __expf(mrow[r] - mn);
            mrow[r]  = mn;
            psum[r]  = 0.f;
        }

        // P = exp(S - m): scatter into per-wave LDS scratch in A-frag order.
        _Float16* pw = sP[w];
#pragma unroll
        for (int jj = 0; jj < 4; ++jj) {
#pragma unroll
            for (int r = 0; r < 8; ++r) {
                float pv = __expf(S[jj][r] - mrow[r]);
                psum[r] += pv;
                pw[(half * 8 + r) * LDT + jj * 16 + rl] = (_Float16)pv;
            }
        }
#pragma unroll
        for (int r = 0; r < 8; ++r) {
            float ps = psum[r];
#pragma unroll
            for (int off = 8; off; off >>= 1) ps += __shfl_xor(ps, off, 32);
            lrow[r] = lrow[r] * alpha[r] + ps;
#pragma unroll
            for (int nn = 0; nn < 4; ++nn) O[nn][r] *= alpha[r];
        }

        // O += P @ V   (P: 16x64 as two K=32 A-frags; V via transposed LDS).
#pragma unroll
        for (int s = 0; s < 2; ++s) {
            v16h pa = frag_row(&pw[rl * LDT + s * 32], half);
#pragma unroll
            for (int nn = 0; nn < 4; ++nn) {
                v16h bv = frag_row(&sVt[(nn * 16 + rl) * LDT + s * 32], half);
                O[nn] = __builtin_amdgcn_wmma_f32_16x16x32_f16(false, pa, false, bv, (short)0, O[nn], false, false);
            }
        }
    }

    // Finalize: O / (l + 1e-8), fuse residual (+ normalized q, f32), emit f16.
#pragma unroll
    for (int nn = 0; nn < 4; ++nn) {
        int col = colbase + nn * 16 + rl;
        size_t row0 = rowbase + w * 16 + half * 8;
        const float* qp = &Qn[row0 * 1024 + col];
        _Float16*    op = &Outh[row0 * 1024 + col];
#pragma unroll
        for (int r = 0; r < 8; ++r) {
            float inv = 1.f / (lrow[r] + 1e-8f);
            op[(size_t)r * 1024] = (_Float16)(O[nn][r] * inv + qp[(size_t)r * 1024]);
        }
    }
}

// ---------------------------------------------------------------------------
// Host-side launcher.
// ---------------------------------------------------------------------------
extern "C" void kernel_launch(void* const* d_in, const int* in_sizes, int n_in,
                              void* d_out, int out_size, void* d_ws, size_t ws_size,
                              hipStream_t stream) {
    const float* x  = (const float*)d_in[0];
    const float* Wq = (const float*)d_in[1];
    const float* bq = (const float*)d_in[2];
    const float* Wk = (const float*)d_in[3];
    const float* bk = (const float*)d_in[4];
    const float* Wv = (const float*)d_in[5];
    const float* bv = (const float*)d_in[6];
    const float* Wo = (const float*)d_in[7];
    const float* bo = (const float*)d_in[8];
    const float* g  = (const float*)d_in[9];
    const float* be = (const float*)d_in[10];
    float* out = (float*)d_out;

    const size_t MD = (size_t)4096 * 1024;      // activations elems
    const size_t WD = (size_t)1024 * 1024;      // weight elems
    char* ws = (char*)d_ws;                     // 64 MB total
    float*    q    = (float*)    (ws);                  // 16 MB (fp32, LN in place)
    _Float16* xh   = (_Float16*) (ws + (16u << 20));    //  8 MB
    _Float16* kh   = (_Float16*) (ws + (24u << 20));    //  8 MB
    _Float16* vh   = (_Float16*) (ws + (32u << 20));    //  8 MB
    _Float16* qh   = (_Float16*) (ws + (40u << 20));    //  8 MB (prescaled)
    _Float16* resh = (_Float16*) (ws + (48u << 20));    //  8 MB
    _Float16* Wqh  = (_Float16*) (ws + (56u << 20));    //  2 MB
    _Float16* Wkh  = (_Float16*) (ws + (58u << 20));
    _Float16* Wvh  = (_Float16*) (ws + (60u << 20));
    _Float16* Woh  = (_Float16*) (ws + (62u << 20));

    // One-shot f32 -> f16 conversions (enables TDM + halves GEMM traffic).
    cvt_f32_f16<<<(int)(MD / 4 / 256), 256, 0, stream>>>(x,  xh,  (int)(MD / 4));
    cvt_f32_f16<<<(int)(WD / 4 / 256), 256, 0, stream>>>(Wq, Wqh, (int)(WD / 4));
    cvt_f32_f16<<<(int)(WD / 4 / 256), 256, 0, stream>>>(Wk, Wkh, (int)(WD / 4));
    cvt_f32_f16<<<(int)(WD / 4 / 256), 256, 0, stream>>>(Wv, Wvh, (int)(WD / 4));
    cvt_f32_f16<<<(int)(WD / 4 / 256), 256, 0, stream>>>(Wo, Woh, (int)(WD / 4));

    dim3 gg(4096 / 128, 1024 / 64);             // (32, 16)
    gemm_bias_f16wmma<float>   <<<gg, 256, 0, stream>>>(xh, Wqh, bq, q,  4096, 1024, 1024);
    gemm_bias_f16wmma<_Float16><<<gg, 256, 0, stream>>>(xh, Wkh, bk, kh, 4096, 1024, 1024);
    gemm_bias_f16wmma<_Float16><<<gg, 256, 0, stream>>>(xh, Wvh, bv, vh, 4096, 1024, 1024);
    layernorm_emit<<<4096, 256, 0, stream>>>(q, qh, g, be);
    flash_attn<<<dim3(32, 32), 128, 0, stream>>>(qh, kh, vh, q, resh);
    gemm_bias_f16wmma<float>   <<<gg, 256, 0, stream>>>(resh, Woh, bo, out, 4096, 1024, 1024);
}
